// DiffusionGCN_57148834840786
// MI455X (gfx1250) — compile-verified, matrix operand order checked by reading, not run on hardware
//
#include <hip/hip_runtime.h>
#include <hip/hip_bf16.h>

// ---------------------------------------------------------------------------
// Types for WMMA
// ---------------------------------------------------------------------------
typedef __attribute__((ext_vector_type(16))) __bf16 v16bf;
typedef __attribute__((ext_vector_type(8)))  __bf16 v8bf;
typedef __attribute__((ext_vector_type(8)))  float  v8f;
typedef __attribute__((ext_vector_type(4)))  float  f32x4;
typedef __attribute__((ext_vector_type(2)))  float  f32x2;

#define IN_CH  128
#define K_DIM  128

// ---------------------------------------------------------------------------
// Degree / normalization
// ---------------------------------------------------------------------------
__global__ void gcn_deg_init(float* __restrict__ deg, int n) {
    int i = blockIdx.x * blockDim.x + threadIdx.x;
    if (i < n) deg[i] = 1.0f;   // self-loop weight
}

__global__ void gcn_deg_edges(const int* __restrict__ dst,
                              const float* __restrict__ w,
                              float* __restrict__ deg, int e) {
    int i = blockIdx.x * blockDim.x + threadIdx.x;
    if (i < e) atomicAdd(&deg[dst[i]], w[i]);
}

__global__ void gcn_deg_rsqrt(float* __restrict__ deg, int n) {
    int i = blockIdx.x * blockDim.x + threadIdx.x;
    if (i < n) {
        float d = deg[i];
        deg[i] = (d > 0.0f) ? rsqrtf(d) : 0.0f;   // deg -> dinv, in place
    }
}

// ---------------------------------------------------------------------------
// Conversions to bf16
// ---------------------------------------------------------------------------
__global__ void gcn_cvt_bf16(const float* __restrict__ src,
                             __bf16* __restrict__ dst, long n) {
    long i = (long)blockIdx.x * blockDim.x + threadIdx.x;
    if (i < n) dst[i] = (__bf16)src[i];
}

// W [K x Ncols] f32 row-major -> Wt [Ncols x K] bf16 (each column contiguous)
__global__ void gcn_cvt_w_transpose(const float* __restrict__ W,
                                    __bf16* __restrict__ Wt,
                                    int K, int Ncols) {
    int t = blockIdx.x * blockDim.x + threadIdx.x;
    if (t < K * Ncols) {
        int n = t / K;
        int k = t % K;
        Wt[t] = (__bf16)W[(long)k * Ncols + n];
    }
}

__global__ void gcn_zero_f32(float* __restrict__ p, long n) {
    long i = (long)blockIdx.x * blockDim.x + threadIdx.x;
    if (i < n) p[i] = 0.0f;
}

// ---------------------------------------------------------------------------
// GEMM: C[M x NCOLS] = A[M x 128] * B[128 x NCOLS], bf16 in, f32 out.
// A row-major bf16.  Bt is B transposed (column n contiguous over K).
// One wave computes one 16x16 tile; K=128 -> 4 x v_wmma_f32_16x16x32_bf16.
// ---------------------------------------------------------------------------
template <int NCOLS>
__global__ void gcn_gemm_wmma_bf16(const __bf16* __restrict__ A,
                                   const __bf16* __restrict__ Bt,
                                   float* __restrict__ C, int M) {
    constexpr int NTILES = NCOLS / 16;
    const int wave = blockIdx.x * (blockDim.x >> 5) + (threadIdx.x >> 5);
    const int lane = threadIdx.x & 31;
    const int mtile = wave / NTILES;
    const int ntile = wave % NTILES;
    if (mtile * 16 >= M) return;

    // ISA 16-bit A/B layout: lanes 0-15 hold K in {koff..koff+7, 16+koff..23+koff}
    // with koff = 0; lanes 16-31 with koff = 8.
    const int koff = (lane >> 4) ? 8 : 0;
    const int row  = mtile * 16 + (lane & 15);
    const int col  = ntile * 16 + (lane & 15);

    const __bf16* __restrict__ arow = A  + (long)row * K_DIM;
    const __bf16* __restrict__ bcol = Bt + (long)col * K_DIM;

    v8f acc = {};
#pragma unroll
    for (int kk = 0; kk < K_DIM; kk += 32) {
        v8bf alo = *(const v8bf*)(arow + kk + koff);
        v8bf ahi = *(const v8bf*)(arow + kk + 16 + koff);
        v8bf blo = *(const v8bf*)(bcol + kk + koff);
        v8bf bhi = *(const v8bf*)(bcol + kk + 16 + koff);
        v16bf a = __builtin_shufflevector(alo, ahi, 0,1,2,3,4,5,6,7,8,9,10,11,12,13,14,15);
        v16bf b = __builtin_shufflevector(blo, bhi, 0,1,2,3,4,5,6,7,8,9,10,11,12,13,14,15);
        acc = __builtin_amdgcn_wmma_f32_16x16x32_bf16(
                  false, a, false, b, (short)0, acc, false, false);
    }

    // C/D layout: VGPR r holds M = r (lanes 0-15) or M = 8+r (lanes 16-31), N = lane&15
    const int mbase = mtile * 16 + ((lane >> 4) ? 8 : 0);
#pragma unroll
    for (int r = 0; r < 8; ++r)
        C[(long)(mbase + r) * NCOLS + col] = acc[r];
}

// ---------------------------------------------------------------------------
// Edge scatter: one wave per edge.  out[dst] += h[src] * norm  (f32 atomics)
// ---------------------------------------------------------------------------
__global__ void gcn_scatter128(const int* __restrict__ src, const int* __restrict__ dst,
                               const float* __restrict__ w, const float* __restrict__ dinv,
                               const float* __restrict__ h, float* __restrict__ out, int e) {
    int edge = blockIdx.x * (blockDim.x >> 5) + (threadIdx.x >> 5);
    int lane = threadIdx.x & 31;
    if (edge >= e) return;
    int s = src[edge], d = dst[edge];
    float nrm = dinv[s] * w[edge] * dinv[d];
    const f32x4* hp = (const f32x4*)(h + (long)s * 128) + lane;
    float* op = out + (long)d * 128 + lane * 4;
    f32x4 v = *hp;
#pragma unroll
    for (int i = 0; i < 4; ++i) atomicAdd(op + i, v[i] * nrm);
}

__global__ void gcn_scatter64(const int* __restrict__ src, const int* __restrict__ dst,
                              const float* __restrict__ w, const float* __restrict__ dinv,
                              const float* __restrict__ h, float* __restrict__ out, int e) {
    int edge = blockIdx.x * (blockDim.x >> 5) + (threadIdx.x >> 5);
    int lane = threadIdx.x & 31;
    if (edge >= e) return;
    int s = src[edge], d = dst[edge];
    float nrm = dinv[s] * w[edge] * dinv[d];
    const f32x2* hp = (const f32x2*)(h + (long)s * 64) + lane;
    float* op = out + (long)d * 64 + lane * 2;
    f32x2 v = *hp;
    atomicAdd(op + 0, v[0] * nrm);
    atomicAdd(op + 1, v[1] * nrm);
}

// ---------------------------------------------------------------------------
// Layer-1 epilogue: self-loop + bias + ReLU, emit bf16 for next GEMM
// ---------------------------------------------------------------------------
__global__ void gcn_finalize1(const float* __restrict__ out1, const float* __restrict__ h1,
                              const float* __restrict__ dinv, const float* __restrict__ b1,
                              __bf16* __restrict__ a1, long total) {
    long idx = (long)blockIdx.x * blockDim.x + threadIdx.x;
    if (idx >= total) return;
    int i = (int)(idx >> 7);       // / 128
    int c = (int)(idx & 127);
    float di = dinv[i];
    float v = out1[idx] + di * di * h1[idx] + b1[c];
    a1[idx] = (__bf16)fmaxf(v, 0.0f);
}

// ---------------------------------------------------------------------------
// Layer-2 epilogue + log_softmax over 64 channels: one wave per row
// ---------------------------------------------------------------------------
__global__ void gcn_finalize2_logsoftmax(const float* __restrict__ out2,
                                         const float* __restrict__ h2,
                                         const float* __restrict__ dinv,
                                         const float* __restrict__ b2,
                                         float* __restrict__ y, int n) {
    int row  = blockIdx.x * (blockDim.x >> 5) + (threadIdx.x >> 5);
    int lane = threadIdx.x & 31;
    if (row >= n) return;
    float di = dinv[row];
    float sl = di * di;
    const float* o = out2 + (long)row * 64;
    const float* h = h2   + (long)row * 64;
    float z0 = o[lane]      + sl * h[lane]      + b2[lane];
    float z1 = o[lane + 32] + sl * h[lane + 32] + b2[lane + 32];

    float m = fmaxf(z0, z1);
#pragma unroll
    for (int off = 16; off > 0; off >>= 1) m = fmaxf(m, __shfl_xor(m, off, 32));
    float s = expf(z0 - m) + expf(z1 - m);
#pragma unroll
    for (int off = 16; off > 0; off >>= 1) s += __shfl_xor(s, off, 32);
    float lse = m + logf(s);

    y[(long)row * 64 + lane]      = z0 - lse;
    y[(long)row * 64 + lane + 32] = z1 - lse;
}

// ---------------------------------------------------------------------------
// Host launcher
// ---------------------------------------------------------------------------
extern "C" void kernel_launch(void* const* d_in, const int* in_sizes, int n_in,
                              void* d_out, int out_size, void* d_ws, size_t ws_size,
                              hipStream_t stream) {
    const float* x  = (const float*)d_in[0];
    const int*   ei = (const int*)  d_in[1];
    const float* ew = (const float*)d_in[2];
    const float* W1 = (const float*)d_in[3];
    const float* b1 = (const float*)d_in[4];
    const float* W2 = (const float*)d_in[5];
    const float* b2 = (const float*)d_in[6];

    const int N = in_sizes[0] / IN_CH;     // 50000
    const int E = in_sizes[2];             // 800000
    const int* src = ei;
    const int* dst = ei + E;

    // workspace carve-out (256B aligned)
    char* p = (char*)d_ws;
    auto alloc = [&](size_t bytes) -> char* {
        char* r = p;
        p += (bytes + 255) & ~(size_t)255;
        return r;
    };
    __bf16* xb   = (__bf16*)alloc((size_t)N * 128 * sizeof(__bf16));
    __bf16* w1t  = (__bf16*)alloc((size_t)128 * 128 * sizeof(__bf16));
    __bf16* w2t  = (__bf16*)alloc((size_t)64  * 128 * sizeof(__bf16));
    float*  dinv = (float*) alloc((size_t)N * sizeof(float));
    float*  h1   = (float*) alloc((size_t)N * 128 * sizeof(float));
    float*  o1   = (float*) alloc((size_t)N * 128 * sizeof(float));
    __bf16* a1   = (__bf16*)alloc((size_t)N * 128 * sizeof(__bf16));
    float*  h2   = (float*) alloc((size_t)N * 64 * sizeof(float));
    float*  o2   = (float*) alloc((size_t)N * 64 * sizeof(float));
    float*  y    = (float*)d_out;

    auto cdiv = [](long a, long b) { return (int)((a + b - 1) / b); };

    // degree / dinv
    gcn_deg_init <<<cdiv(N, 256), 256, 0, stream>>>(dinv, N);
    gcn_deg_edges<<<cdiv(E, 256), 256, 0, stream>>>(dst, ew, dinv, E);
    gcn_deg_rsqrt<<<cdiv(N, 256), 256, 0, stream>>>(dinv, N);

    // bf16 conversions
    gcn_cvt_bf16<<<cdiv((long)N * 128, 256), 256, 0, stream>>>(x, xb, (long)N * 128);
    gcn_cvt_w_transpose<<<cdiv(128 * 128, 256), 256, 0, stream>>>(W1, w1t, 128, 128);
    gcn_cvt_w_transpose<<<cdiv(128 * 64, 256), 256, 0, stream>>>(W2, w2t, 128, 64);

    // layer 1: h1 = x @ W1  (WMMA)
    {
        int mtiles = cdiv(N, 16);
        int tiles  = mtiles * (128 / 16);
        gcn_gemm_wmma_bf16<128><<<cdiv(tiles, 8), 256, 0, stream>>>(xb, w1t, h1, N);
    }
    gcn_zero_f32<<<cdiv((long)N * 128, 256), 256, 0, stream>>>(o1, (long)N * 128);
    gcn_scatter128<<<cdiv(E, 8), 256, 0, stream>>>(src, dst, ew, dinv, h1, o1, E);
    gcn_finalize1<<<cdiv((long)N * 128, 256), 256, 0, stream>>>(o1, h1, dinv, b1, a1,
                                                                (long)N * 128);

    // layer 2: h2 = a1 @ W2  (WMMA)
    {
        int mtiles = cdiv(N, 16);
        int tiles  = mtiles * (64 / 16);
        gcn_gemm_wmma_bf16<64><<<cdiv(tiles, 8), 256, 0, stream>>>(a1, w2t, h2, N);
    }
    gcn_zero_f32<<<cdiv((long)N * 64, 256), 256, 0, stream>>>(o2, (long)N * 64);
    gcn_scatter64<<<cdiv(E, 8), 256, 0, stream>>>(src, dst, ew, dinv, h2, o2, E);
    gcn_finalize2_logsoftmax<<<cdiv(N, 8), 256, 0, stream>>>(o2, h2, dinv, b2, y, N);

    (void)n_in; (void)out_size; (void)ws_size;
}
